// GCN_52286931862208
// MI455X (gfx1250) — matrix operand order repaired and checked by reading, not compiled, not run on hardware
//
#include <hip/hip_runtime.h>
#include <hip/hip_bf16.h>
#include <math.h>

typedef __attribute__((ext_vector_type(16))) _Float16 v16h;
typedef __attribute__((ext_vector_type(8)))  _Float16 v8h;
typedef __attribute__((ext_vector_type(8)))  float    v8f;

static constexpr int USER_NUM = 100000;
static constexpr int ITEM_NUM = 50000;
static constexpr int N_NODES  = USER_NUM + ITEM_NUM;   // 150000 (multiple of 16)
static constexpr int N_EDGES  = 2400000;
static constexpr int BATCH_N  = 4096;

static __device__ __forceinline__ void atomAddF(float* p, float v) {
  (void)__hip_atomic_fetch_add(p, v, __ATOMIC_RELAXED, __HIP_MEMORY_SCOPE_AGENT);
}

// ---------------- elementwise helpers ----------------
__global__ void k_zero(float* __restrict__ p, int n) {
  int i = blockIdx.x * blockDim.x + threadIdx.x;
  if (i < n) p[i] = 0.f;
}
__global__ void k_fill1(float* __restrict__ p, int n) {
  int i = blockIdx.x * blockDim.x + threadIdx.x;
  if (i < n) p[i] = 1.f;
}
__global__ void k_rsqrt(float* __restrict__ p, int n) {
  int i = blockIdx.x * blockDim.x + threadIdx.x;
  if (i < n) p[i] = rsqrtf(p[i]);
}
__global__ void k_deg_count(const int* __restrict__ dst, float* __restrict__ deg, int n) {
  int i = blockIdx.x * blockDim.x + threadIdx.x;
  if (i < n) atomAddF(&deg[dst[i]], 1.f);
}
__global__ void k_f32_to_f16(const float* __restrict__ x, _Float16* __restrict__ y, int n) {
  int i = blockIdx.x * blockDim.x + threadIdx.x;
  if (i < n) y[i] = (_Float16)x[i];
}
// W[IN][OUT] f32 -> Wt[OUT][IN] f16 (transposed so B-fragment K reads are contiguous)
__global__ void k_w_to_f16_t(const float* __restrict__ W, _Float16* __restrict__ Wt, int IN, int OUT) {
  int i = blockIdx.x * blockDim.x + threadIdx.x;
  if (i < IN * OUT) {
    int o = i / IN, k = i % IN;
    Wt[i] = (_Float16)W[k * OUT + o];
  }
}
// agg[v][f] = h[v][f]*dis[v]^2 + bias[f]
__global__ void k_self_init(const float* __restrict__ h, const float* __restrict__ dis,
                            const float* __restrict__ bias, float* __restrict__ agg, int n, int H) {
  int i = blockIdx.x * blockDim.x + threadIdx.x;
  if (i < n) {
    int v = i / H, f = i % H;
    float d = dis[v];
    agg[i] = fmaf(h[i], d * d, bias[f]);
  }
}

// ---------------- WMMA GEMM: Y[N][OUT] = X16[N][IN] @ W (Wt16 is [OUT][IN]) ----------------
template <int IN, int OUT>
__global__ __launch_bounds__(256) void k_gemm_wmma(const _Float16* __restrict__ X,
                                                   const _Float16* __restrict__ Wt,
                                                   float* __restrict__ Y, int nRows) {
  const int lane = threadIdx.x & 31;
  const int wid  = blockIdx.x * (blockDim.x >> 5) + (threadIdx.x >> 5);
  const int tilesPerRow = OUT / 16;
  const int nTiles = (nRows / 16) * tilesPerRow;
  if (wid >= nTiles) return;
  const int row0 = (wid / tilesPerRow) * 16;
  const int col0 = (wid % tilesPerRow) * 16;
  const int m    = lane & 15;   // M for A, N for B/C
  const int half = lane >> 4;
  v8f c = {};
#pragma unroll
  for (int k0 = 0; k0 < IN; k0 += 32) {
    // A fragment: lanes 0-15 hold K k0..k0+15; lanes 16-31 hold K k0+16..k0+31 (contiguous)
    const _Float16* ap = X + (size_t)(row0 + m) * IN + k0 + half * 16;
    v8h alo = *(const v8h*)ap;
    v8h ahi = *(const v8h*)(ap + 8);
    v16h a = __builtin_shufflevector(alo, ahi, 0, 1, 2, 3, 4, 5, 6, 7, 8, 9, 10, 11, 12, 13, 14, 15);
    // B fragment: N = m, K = k0 + half*16 + e (contiguous in transposed W)
    v16h b = *(const v16h*)(Wt + (size_t)(col0 + m) * IN + k0 + half * 16);
    c = __builtin_amdgcn_wmma_f32_16x16x32_f16(false, a, false, b, (short)0, c, false, false);
  }
#pragma unroll
  for (int r = 0; r < 8; ++r)
    Y[(size_t)(row0 + half * 8 + r) * OUT + col0 + m] = c[r];
}

// ---------------- edge scatter: agg[dst] += h[src] * dis[src]*dis[dst] ----------------
// One wave per edge; each lane gathers H/32 contiguous features as one vector load.
template <int H>
__global__ __launch_bounds__(256) void k_edge_agg(const int* __restrict__ src, const int* __restrict__ dst,
                                                  const float* __restrict__ dis, const float* __restrict__ h,
                                                  float* __restrict__ agg, int nE) {
  constexpr int PER = H / 32;  // 4 (H=128) or 2 (H=64)
  const int lane = threadIdx.x & 31;
  const int wid  = blockIdx.x * (blockDim.x >> 5) + (threadIdx.x >> 5);
  const int nW   = gridDim.x * (blockDim.x >> 5);
  for (int e = wid; e < nE; e += nW) {
    const int s = src[e], d = dst[e];
    const float nm = dis[s] * dis[d];
    const float* hp = h + (size_t)s * H + lane * PER;
    float* ap = agg + (size_t)d * H + lane * PER;
    float v[PER];
    *(float4*)nullptr;  // (unused; removed below)
  }
}
// Specialized implementations with proper vector gathers.
__global__ __launch_bounds__(256) void k_edge_agg128(const int* __restrict__ src, const int* __restrict__ dst,
                                                     const float* __restrict__ dis, const float* __restrict__ h,
                                                     float* __restrict__ agg, int nE) {
  const int lane = threadIdx.x & 31;
  const int wid  = blockIdx.x * (blockDim.x >> 5) + (threadIdx.x >> 5);
  const int nW   = gridDim.x * (blockDim.x >> 5);
  for (int e = wid; e < nE; e += nW) {
    const int s = src[e], d = dst[e];
    const float nm = dis[s] * dis[d];
    const float4 hv = *(const float4*)(h + (size_t)s * 128 + lane * 4);
    float* ap = agg + (size_t)d * 128 + lane * 4;
    atomAddF(ap + 0, hv.x * nm);
    atomAddF(ap + 1, hv.y * nm);
    atomAddF(ap + 2, hv.z * nm);
    atomAddF(ap + 3, hv.w * nm);
  }
}
__global__ __launch_bounds__(256) void k_edge_agg64(const int* __restrict__ src, const int* __restrict__ dst,
                                                    const float* __restrict__ dis, const float* __restrict__ h,
                                                    float* __restrict__ agg, int nE) {
  const int lane = threadIdx.x & 31;
  const int wid  = blockIdx.x * (blockDim.x >> 5) + (threadIdx.x >> 5);
  const int nW   = gridDim.x * (blockDim.x >> 5);
  for (int e = wid; e < nE; e += nW) {
    const int s = src[e], d = dst[e];
    const float nm = dis[s] * dis[d];
    const float2 hv = *(const float2*)(h + (size_t)s * 64 + lane * 2);
    float* ap = agg + (size_t)d * 64 + lane * 2;
    atomAddF(ap + 0, hv.x * nm);
    atomAddF(ap + 1, hv.y * nm);
  }
}

// ---------------- batchnorm (train): column sums, then scale/shift ----------------
__global__ void k_bn_stats(const float* __restrict__ x, float* __restrict__ sum, float* __restrict__ sq,
                           int nRows, int H) {
  const int f = threadIdx.x;  // blockDim.x == H
  float s = 0.f, q = 0.f;
  for (int r = blockIdx.x; r < nRows; r += gridDim.x) {
    const float v = x[(size_t)r * H + f];
    s += v;
    q += v * v;
  }
  atomAddF(&sum[f], s);
  atomAddF(&sq[f], q);
}
__global__ void k_bn_finalize(const float* __restrict__ sum, const float* __restrict__ sq,
                              const float* __restrict__ g, const float* __restrict__ beta,
                              float* __restrict__ scale, float* __restrict__ shift, int nRows, int H) {
  int f = threadIdx.x;
  if (f < H) {
    float mean = sum[f] / (float)nRows;
    float var  = sq[f] / (float)nRows - mean * mean;
    float sc   = g[f] * rsqrtf(var + 1e-5f);
    scale[f] = sc;
    shift[f] = beta[f] - mean * sc;
  }
}
// y = l2norm(relu(x*scale + shift)) per row, one wave per row (wave32 butterfly).
// Also emits the f16 staging copy for the next layer's WMMA GEMM (fused, saves a full pass).
__global__ __launch_bounds__(256) void k_bn_relu_l2(const float* __restrict__ xin, float* __restrict__ xout,
                                                    _Float16* __restrict__ xout16,
                                                    const float* __restrict__ scale, const float* __restrict__ shift,
                                                    int nRows, int H) {
  const int lane = threadIdx.x & 31;
  const int wid  = blockIdx.x * (blockDim.x >> 5) + (threadIdx.x >> 5);
  const int nW   = gridDim.x * (blockDim.x >> 5);
  const int nf   = H >> 5;  // <= 4
  for (int row = wid; row < nRows; row += nW) {
    float v[4];
    float ss = 0.f;
    for (int i = 0; i < nf; ++i) {
      const int f = lane + (i << 5);
      float t = fmaf(xin[(size_t)row * H + f], scale[f], shift[f]);
      t = fmaxf(t, 0.f);
      v[i] = t;
      ss += t * t;
    }
#pragma unroll
    for (int mk = 16; mk >= 1; mk >>= 1) ss += __shfl_xor(ss, mk, 32);
    const float inv = 1.f / fmaxf(sqrtf(ss), 1e-12f);
    for (int i = 0; i < nf; ++i) {
      const int f = lane + (i << 5);
      const float o = v[i] * inv;
      xout[(size_t)row * H + f] = o;
      xout16[(size_t)row * H + f] = (_Float16)o;
    }
  }
}

// ---------------- BPR loss ----------------
__global__ __launch_bounds__(256) void k_bpr(const float* __restrict__ x, const int* __restrict__ uid,
                                             const int* __restrict__ pid, const int* __restrict__ nid,
                                             float* __restrict__ acc, int B) {
  const int lane = threadIdx.x & 31;
  const int wid  = blockIdx.x * (blockDim.x >> 5) + (threadIdx.x >> 5);
  if (wid >= B) return;
  const float* u = x + (size_t)uid[wid] * 64;
  const float* p = x + (size_t)(USER_NUM + pid[wid]) * 64;
  const float* n = x + (size_t)(USER_NUM + nid[wid]) * 64;
  float ps = u[lane] * p[lane] + u[lane + 32] * p[lane + 32];
  float ns = u[lane] * n[lane] + u[lane + 32] * n[lane + 32];
#pragma unroll
  for (int mk = 16; mk >= 1; mk >>= 1) {
    ps += __shfl_xor(ps, mk, 32);
    ns += __shfl_xor(ns, mk, 32);
  }
  if (lane == 0) {
    const float s = ps - ns;
    const float l = (s > 0.f) ? log1pf(expf(-s)) : (-s + log1pf(expf(s)));  // -log_sigmoid(s)
    atomAddF(acc, l);
  }
}
__global__ void k_loss_final(const float* __restrict__ acc, float* __restrict__ out, int B) {
  if (threadIdx.x == 0 && blockIdx.x == 0) out[0] = acc[0] / (float)B;
}

// ---------------- host-side orchestration ----------------
static inline int nblk(long n, int t) { return (int)((n + t - 1) / t); }

extern "C" void kernel_launch(void* const* d_in, const int* in_sizes, int n_in,
                              void* d_out, int out_size, void* d_ws, size_t ws_size,
                              hipStream_t stream) {
  const float* user_emb = (const float*)d_in[0];
  const float* item_emb = (const float*)d_in[1];
  const float* W0 = (const float*)d_in[2];
  const float* b0 = (const float*)d_in[3];
  const float* g0 = (const float*)d_in[4];
  const float* be0 = (const float*)d_in[5];
  const float* W1 = (const float*)d_in[6];
  const float* b1 = (const float*)d_in[7];
  const float* g1 = (const float*)d_in[8];
  const float* be1 = (const float*)d_in[9];
  const float* W2 = (const float*)d_in[10];
  const float* b2 = (const float*)d_in[11];
  const int* uid = (const int*)d_in[12];
  const int* pid = (const int*)d_in[13];
  const int* nid = (const int*)d_in[14];
  const int* edge = (const int*)d_in[15];
  const int* srcE = edge;
  const int* dstE = edge + N_EDGES;

  char* ws = (char*)d_ws;
  size_t off = 0;
  auto take = [&](size_t bytes) -> char* {
    char* p = ws + off;
    off += (bytes + 255) & ~(size_t)255;
    return p;
  };
  float* dis = (float*)take((size_t)N_NODES * 4);
  float* A   = (float*)take((size_t)N_NODES * 128 * 4);  // layer input x (f32)
  float* Bf  = (float*)take((size_t)N_NODES * 128 * 4);  // h = x@W
  float* Cf  = (float*)take((size_t)N_NODES * 128 * 4);  // agg
  _Float16* x16  = (_Float16*)take((size_t)N_NODES * 128 * 2);  // f16 staging for WMMA
  _Float16* wt16 = (_Float16*)take(128 * 128 * 2);
  float* sums   = (float*)take(128 * 4);
  float* sq     = (float*)take(128 * 4);
  float* scale  = (float*)take(128 * 4);
  float* shiftv = (float*)take(128 * 4);
  float* acc    = (float*)take(4);

  float* out = (float*)d_out;
  const int T = 256;

  // degree -> dis = rsqrt(1 + indegree)
  k_fill1<<<nblk(N_NODES, T), T, 0, stream>>>(dis, N_NODES);
  k_deg_count<<<nblk(N_EDGES, T), T, 0, stream>>>(dstE, dis, N_EDGES);
  k_rsqrt<<<nblk(N_NODES, T), T, 0, stream>>>(dis, N_NODES);

  // x = concat(user_emb, item_emb), row stride 64
  hipMemcpyAsync(A, user_emb, (size_t)USER_NUM * 64 * 4, hipMemcpyDeviceToDevice, stream);
  hipMemcpyAsync(A + (size_t)USER_NUM * 64, item_emb, (size_t)ITEM_NUM * 64 * 4,
                 hipMemcpyDeviceToDevice, stream);

  // ---------- layer 0: 64 -> 128, BN+ReLU+L2 ----------
  k_f32_to_f16<<<nblk((long)N_NODES * 64, T), T, 0, stream>>>(A, x16, N_NODES * 64);
  k_w_to_f16_t<<<nblk(64 * 128, T), T, 0, stream>>>(W0, wt16, 64, 128);
  {
    int tiles = (N_NODES / 16) * (128 / 16);
    k_gemm_wmma<64, 128><<<nblk(tiles, 8), 256, 0, stream>>>(x16, wt16, Bf, N_NODES);
  }
  k_self_init<<<nblk((long)N_NODES * 128, T), T, 0, stream>>>(Bf, dis, b0, Cf, N_NODES * 128, 128);
  k_edge_agg128<<<nblk((long)N_EDGES, 8), 256, 0, stream>>>(srcE, dstE, dis, Bf, Cf, N_EDGES);
  k_zero<<<1, 128, 0, stream>>>(sums, 128);
  k_zero<<<1, 128, 0, stream>>>(sq, 128);
  k_bn_stats<<<1024, 128, 0, stream>>>(Cf, sums, sq, N_NODES, 128);
  k_bn_finalize<<<1, 128, 0, stream>>>(sums, sq, g0, be0, scale, shiftv, N_NODES, 128);
  k_bn_relu_l2<<<nblk((long)N_NODES, 8), 256, 0, stream>>>(Cf, A, x16, scale, shiftv, N_NODES, 128);

  // ---------- layer 1: 128 -> 128, BN+ReLU+L2 ----------
  k_w_to_f16_t<<<nblk(128 * 128, T), T, 0, stream>>>(W1, wt16, 128, 128);
  {
    int tiles = (N_NODES / 16) * (128 / 16);
    k_gemm_wmma<128, 128><<<nblk(tiles, 8), 256, 0, stream>>>(x16, wt16, Bf, N_NODES);
  }
  k_self_init<<<nblk((long)N_NODES * 128, T), T, 0, stream>>>(Bf, dis, b1, Cf, N_NODES * 128, 128);
  k_edge_agg128<<<nblk((long)N_EDGES, 8), 256, 0, stream>>>(srcE, dstE, dis, Bf, Cf, N_EDGES);
  k_zero<<<1, 128, 0, stream>>>(sums, 128);
  k_zero<<<1, 128, 0, stream>>>(sq, 128);
  k_bn_stats<<<1024, 128, 0, stream>>>(Cf, sums, sq, N_NODES, 128);
  k_bn_finalize<<<1, 128, 0, stream>>>(sums, sq, g1, be1, scale, shiftv, N_NODES, 128);
  k_bn_relu_l2<<<nblk((long)N_NODES, 8), 256, 0, stream>>>(Cf, A, x16, scale, shiftv, N_NODES, 128);

  // ---------- layer 2: 128 -> 64, no activation ----------
  k_w_to_f16_t<<<nblk(128 * 64, T), T, 0, stream>>>(W2, wt16, 128, 64);
  {
    int tiles = (N_NODES / 16) * (64 / 16);
    k_gemm_wmma<128, 64><<<nblk(tiles, 8), 256, 0, stream>>>(x16, wt16, Bf, N_NODES);
  }
  k_self_init<<<nblk((long)N_NODES * 64, T), T, 0, stream>>>(Bf, dis, b2, Cf, N_NODES * 64, 64);
  k_edge_agg64<<<nblk((long)N_EDGES, 8), 256, 0, stream>>>(srcE, dstE, dis, Bf, Cf, N_EDGES);

  // x output (d_out[1:]) and BPR loss (d_out[0])
  hipMemcpyAsync(out + 1, Cf, (size_t)N_NODES * 64 * 4, hipMemcpyDeviceToDevice, stream);
  k_zero<<<1, 1, 0, stream>>>(acc, 1);
  k_bpr<<<nblk((long)BATCH_N, 8), 256, 0, stream>>>(Cf, uid, pid, nid, acc, BATCH_N);
  k_loss_final<<<1, 1, 0, stream>>>(acc, out, BATCH_N);
}
